// MultiHeadAttention_4595615006962
// MI455X (gfx1250) — compile-verified, matrix operand order checked by reading, not run on hardware
//
#include <hip/hip_runtime.h>
#include <hip/hip_bf16.h>
#include <stdint.h>

// ---------------------------------------------------------------------------
// MultiHeadAttention for MI455X (gfx1250): bf16 WMMA everywhere.
// ---------------------------------------------------------------------------

typedef __attribute__((ext_vector_type(16))) __bf16 v16bf_t;
typedef __attribute__((ext_vector_type(8)))  float  v8f_t;

union FragBF { v16bf_t v; uint32_t u[8]; uint16_t s[16]; };
union FragF  { v8f_t   v; float    f[8]; };

__device__ __forceinline__ uint16_t f32_to_bf16(float f) {
  uint32_t u = __float_as_uint(f);
  u += 0x7FFFu + ((u >> 16) & 1u);           // round-to-nearest-even
  return (uint16_t)(u >> 16);
}

#define GK 1024
#define GN 1024
#define BM 128
#define BN 64
#define BK 32
#define LDA 40   // padded bf16 stride (80B rows, 16B aligned, conflict-breaking)
#define LDB 40

// ---------------------------------------------------------------------------
// Generic WMMA GEMM (double-buffered LDS): Out[8192,1024] = A x W
//   A_IS_BF16: 0 -> A is f32 (convert inline), 1 -> A is bf16
//   OUT_MODE : 0 -> bf16, de-interleave col=(d*16+h) into [b][h][s][64]
//              2 -> bf16, de-interleave into [b][h][64][s]   (V transposed)
//              3 -> f32 plain row-major
// ---------------------------------------------------------------------------
template<int A_IS_BF16, int OUT_MODE>
__global__ __launch_bounds__(256)
void gemm_wmma(const void* __restrict__ Aptr, const float* __restrict__ W,
               void* __restrict__ Outp)
{
  __shared__ uint16_t As[2][BM][LDA];
  __shared__ uint16_t Bs[2][BN][LDB];   // stored transposed: Bs[.][n][k]

  const int tid   = threadIdx.x;
  const int wave  = tid >> 5;
  const int lane  = tid & 31;
  const int laneM = lane & 15;
  const int hi    = lane >> 4;
  const int mSub  = (wave & 3) * 32;
  const int nSub  = (wave >> 2) * 32;
  const int blockM = blockIdx.x * BM;
  const int blockN = blockIdx.y * BN;

  FragF acc[2][2];
#pragma unroll
  for (int i = 0; i < 2; i++)
#pragma unroll
    for (int j = 0; j < 2; j++)
#pragma unroll
      for (int r = 0; r < 8; r++) acc[i][j].f[r] = 0.f;

  const int arow  = tid >> 1;          // 0..127
  const int akseg = (tid & 1) * 16;    // 0 or 16
  const int bk    = tid & 31;          // 0..31
  const int bnseg = (tid >> 5) * 8;    // 0..56

  auto stage = [&](int buf, int k0) {
    // ---- stage A tile (convert f32->bf16 if needed) ----
    union { uint16_t s[16]; uint4 q[2]; } tmp;
    if (A_IS_BF16) {
      const uint4* src = (const uint4*)((const uint16_t*)Aptr +
                           (size_t)(blockM + arow) * GK + k0 + akseg);
      tmp.q[0] = src[0]; tmp.q[1] = src[1];
    } else {
      const float4* src = (const float4*)((const float*)Aptr +
                            (size_t)(blockM + arow) * GK + k0 + akseg);
#pragma unroll
      for (int v = 0; v < 4; v++) {
        float4 f = src[v];
        tmp.s[v*4+0] = f32_to_bf16(f.x); tmp.s[v*4+1] = f32_to_bf16(f.y);
        tmp.s[v*4+2] = f32_to_bf16(f.z); tmp.s[v*4+3] = f32_to_bf16(f.w);
      }
    }
    uint4* dst = (uint4*)&As[buf][arow][akseg];
    dst[0] = tmp.q[0]; dst[1] = tmp.q[1];
    // ---- stage B tile transposed: Bs[n][k] ----
    const float4* src = (const float4*)(W + (size_t)(k0 + bk) * GN + blockN + bnseg);
    float4 f0 = src[0], f1 = src[1];
    Bs[buf][bnseg+0][bk] = f32_to_bf16(f0.x); Bs[buf][bnseg+1][bk] = f32_to_bf16(f0.y);
    Bs[buf][bnseg+2][bk] = f32_to_bf16(f0.z); Bs[buf][bnseg+3][bk] = f32_to_bf16(f0.w);
    Bs[buf][bnseg+4][bk] = f32_to_bf16(f1.x); Bs[buf][bnseg+5][bk] = f32_to_bf16(f1.y);
    Bs[buf][bnseg+6][bk] = f32_to_bf16(f1.z); Bs[buf][bnseg+7][bk] = f32_to_bf16(f1.w);
  };

  stage(0, 0);

  for (int k0 = 0; k0 < GK; k0 += BK) {
    const int cur = (k0 >> 5) & 1;
    __syncthreads();                     // tile `cur` ready; prev reads done
    if (k0 + BK < GK) stage(cur ^ 1, k0 + BK);

    // ---- fragments per ISA layout: A lane=row L%16, K words {0..3,8..11}+hi*4
    FragBF afr[2], bfr[2];
#pragma unroll
    for (int mt = 0; mt < 2; mt++) {
      const uint32_t* wp = (const uint32_t*)&As[cur][mSub + mt*16 + laneM][0];
      const int wo = hi * 4;
      afr[mt].u[0]=wp[wo+0]; afr[mt].u[1]=wp[wo+1];
      afr[mt].u[2]=wp[wo+2]; afr[mt].u[3]=wp[wo+3];
      afr[mt].u[4]=wp[wo+8]; afr[mt].u[5]=wp[wo+9];
      afr[mt].u[6]=wp[wo+10]; afr[mt].u[7]=wp[wo+11];
    }
#pragma unroll
    for (int nt = 0; nt < 2; nt++) {
      const uint32_t* wp = (const uint32_t*)&Bs[cur][nSub + nt*16 + laneM][0];
      const int wo = hi * 4;
      bfr[nt].u[0]=wp[wo+0]; bfr[nt].u[1]=wp[wo+1];
      bfr[nt].u[2]=wp[wo+2]; bfr[nt].u[3]=wp[wo+3];
      bfr[nt].u[4]=wp[wo+8]; bfr[nt].u[5]=wp[wo+9];
      bfr[nt].u[6]=wp[wo+10]; bfr[nt].u[7]=wp[wo+11];
    }
#pragma unroll
    for (int mt = 0; mt < 2; mt++)
#pragma unroll
      for (int nt = 0; nt < 2; nt++)
        acc[mt][nt].v = __builtin_amdgcn_wmma_f32_16x16x32_bf16(
            false, afr[mt].v, false, bfr[nt].v,
            (short)0, acc[mt][nt].v, false, false);
  }

  // ---- epilogue (C/D layout: lane = n + 16*(m/8), reg = m%8) ----
#pragma unroll
  for (int mt = 0; mt < 2; mt++)
#pragma unroll
    for (int nt = 0; nt < 2; nt++)
#pragma unroll
      for (int r = 0; r < 8; r++) {
        const int gM = blockM + mSub + mt*16 + r + 8*hi;
        const int gN = blockN + nSub + nt*16 + laneM;
        const float val = acc[mt][nt].f[r];
        if (OUT_MODE == 3) {
          ((float*)Outp)[(size_t)gM * GN + gN] = val;
        } else {
          const int bb = gM >> 11, ss = gM & 2047;   // batch of 2048 rows
          const int dd = gN >> 4,  hh = gN & 15;     // col = d*16 + h
          const uint16_t bv = f32_to_bf16(val);
          if (OUT_MODE == 0)
            ((uint16_t*)Outp)[(((size_t)(bb*16 + hh) * 2048) + ss) * 64 + dd] = bv;
          else
            ((uint16_t*)Outp)[(((size_t)(bb*16 + hh) * 64) + dd) * 2048 + ss] = bv;
        }
      }
}

// ---------------------------------------------------------------------------
// Flash attention: one wave32 per (b, h, 16-query tile); 64-key chunks so the
// cross-lane softmax reductions (ds_bpermute) amortize over 16 WMMAs/iter.
// ---------------------------------------------------------------------------
__global__ __launch_bounds__(32)
void attn_wmma(const uint16_t* __restrict__ Qp, const uint16_t* __restrict__ Kp,
               const uint16_t* __restrict__ VpT, const int* __restrict__ Mask,
               uint16_t* __restrict__ Mha)
{
  __shared__ uint16_t Plds[16][80];   // 16q x 64k P staging (wave-private)

  const int lane  = threadIdx.x;
  const int laneM = lane & 15;
  const int hi    = lane >> 4;
  const int qBase = blockIdx.x * 16;
  const int h     = blockIdx.y;
  const int b     = blockIdx.z;
  const size_t headOff = (size_t)(b * 16 + h);

  const float SCL = 0.125f * 1.44269504088896341f;   // 1/sqrt(64) * log2(e)

  // Q A-fragments (lane holds query row laneM), two K=32 steps over d=64
  FragBF aq[2];
  {
    const uint32_t* wp = (const uint32_t*)(Qp + (headOff*2048 + qBase + laneM) * 64);
#pragma unroll
    for (int ds = 0; ds < 2; ds++) {
      const int wo = hi*4 + ds*16;
      aq[ds].u[0]=wp[wo+0]; aq[ds].u[1]=wp[wo+1];
      aq[ds].u[2]=wp[wo+2]; aq[ds].u[3]=wp[wo+3];
      aq[ds].u[4]=wp[wo+8]; aq[ds].u[5]=wp[wo+9];
      aq[ds].u[6]=wp[wo+10]; aq[ds].u[7]=wp[wo+11];
    }
  }

  float mrow[8], lrow[8];
  FragF oacc[4];
#pragma unroll
  for (int r = 0; r < 8; r++) { mrow[r] = -1e30f; lrow[r] = 0.f; }
#pragma unroll
  for (int nt = 0; nt < 4; nt++)
#pragma unroll
    for (int r = 0; r < 8; r++) oacc[nt].f[r] = 0.f;

  const int* maskBase = Mask + ((size_t)b * 2048 + qBase) * 2048;

  for (int kc = 0; kc < 2048; kc += 64) {
    // ---- S = Q K^T for 16q x 64k ----
    FragF s[4];
#pragma unroll
    for (int kj = 0; kj < 4; kj++) {
#pragma unroll
      for (int r = 0; r < 8; r++) s[kj].f[r] = 0.f;
      const uint32_t* wp =
          (const uint32_t*)(Kp + (headOff*2048 + kc + kj*16 + laneM) * 64);
#pragma unroll
      for (int ds = 0; ds < 2; ds++) {
        FragBF bkf;
        const int wo = hi*4 + ds*16;
        bkf.u[0]=wp[wo+0]; bkf.u[1]=wp[wo+1]; bkf.u[2]=wp[wo+2]; bkf.u[3]=wp[wo+3];
        bkf.u[4]=wp[wo+8]; bkf.u[5]=wp[wo+9]; bkf.u[6]=wp[wo+10]; bkf.u[7]=wp[wo+11];
        s[kj].v = __builtin_amdgcn_wmma_f32_16x16x32_bf16(
            false, aq[ds].v, false, bkf.v, (short)0, s[kj].v, false, false);
      }
    }
    // prefetch next K/V chunks toward the WGP while we crunch this one
    if (kc + 64 < 2048) {
      __builtin_prefetch(Kp + (headOff*2048 + kc + 64 + lane) * 64, 0, 1);
      __builtin_prefetch(VpT + (headOff*64 + (lane & 31)) * 2048 + kc + 64, 0, 1);
    }
    // ---- mask + scale (log2 domain) ----
#pragma unroll
    for (int kj = 0; kj < 4; kj++)
#pragma unroll
      for (int r = 0; r < 8; r++) {
        const int mk = maskBase[(size_t)(r + 8*hi) * 2048 + kc + kj*16 + laneM];
        const float v = s[kj].f[r] * SCL;
        s[kj].f[r] = (mk == 0) ? -1e30f : v;
      }
    // ---- online softmax: row reductions over 16-lane halves ----
#pragma unroll
    for (int r = 0; r < 8; r++) {
      float rm = fmaxf(fmaxf(s[0].f[r], s[1].f[r]),
                       fmaxf(s[2].f[r], s[3].f[r]));
      rm = fmaxf(rm, __shfl_xor(rm, 1));
      rm = fmaxf(rm, __shfl_xor(rm, 2));
      rm = fmaxf(rm, __shfl_xor(rm, 4));
      rm = fmaxf(rm, __shfl_xor(rm, 8));
      const float mnew  = fmaxf(mrow[r], rm);
      const float alpha = exp2f(mrow[r] - mnew);
      mrow[r] = mnew;
      float rs = 0.f;
#pragma unroll
      for (int kj = 0; kj < 4; kj++) {
        const float p = exp2f(s[kj].f[r] - mnew);
        s[kj].f[r] = p;
        rs += p;
      }
      rs += __shfl_xor(rs, 1);
      rs += __shfl_xor(rs, 2);
      rs += __shfl_xor(rs, 4);
      rs += __shfl_xor(rs, 8);
      lrow[r] = lrow[r] * alpha + rs;
#pragma unroll
      for (int nt = 0; nt < 4; nt++) oacc[nt].f[r] *= alpha;
    }
    // ---- P: D-layout -> A-layout transpose through LDS ----
    asm volatile("" ::: "memory");
#pragma unroll
    for (int kj = 0; kj < 4; kj++)
#pragma unroll
      for (int r = 0; r < 8; r++)
        Plds[r + 8*hi][kj*16 + laneM] = f32_to_bf16(s[kj].f[r]);
    asm volatile("s_wait_dscnt 0" ::: "memory");   // same-wave DS RAW
    FragBF ap[2];
#pragma unroll
    for (int kk = 0; kk < 2; kk++) {
      const uint32_t* wp = (const uint32_t*)&Plds[laneM][kk*32];
      const int wo = hi * 4;
      ap[kk].u[0]=wp[wo+0]; ap[kk].u[1]=wp[wo+1];
      ap[kk].u[2]=wp[wo+2]; ap[kk].u[3]=wp[wo+3];
      ap[kk].u[4]=wp[wo+8]; ap[kk].u[5]=wp[wo+9];
      ap[kk].u[6]=wp[wo+10]; ap[kk].u[7]=wp[wo+11];
    }
    asm volatile("" ::: "memory");
    // ---- O += P V  (V^T rows give contiguous keys) ----
#pragma unroll
    for (int nt = 0; nt < 4; nt++) {
      const uint32_t* wp =
          (const uint32_t*)(VpT + (headOff*64 + nt*16 + laneM) * 2048 + kc);
#pragma unroll
      for (int kk = 0; kk < 2; kk++) {
        FragBF bvf;
        const int wo = kk*16 + hi*4;
        bvf.u[0]=wp[wo+0]; bvf.u[1]=wp[wo+1]; bvf.u[2]=wp[wo+2]; bvf.u[3]=wp[wo+3];
        bvf.u[4]=wp[wo+8]; bvf.u[5]=wp[wo+9]; bvf.u[6]=wp[wo+10]; bvf.u[7]=wp[wo+11];
        oacc[nt].v = __builtin_amdgcn_wmma_f32_16x16x32_bf16(
            false, ap[kk].v, false, bvf.v, (short)0, oacc[nt].v, false, false);
      }
    }
  }

  // ---- normalize, write back interleaved (col = d*16 + h) bf16 ----
#pragma unroll
  for (int r = 0; r < 8; r++) {
    const float inv = 1.0f / lrow[r];
    const int q = qBase + r + 8*hi;
#pragma unroll
    for (int nt = 0; nt < 4; nt++) {
      const int c = (nt*16 + laneM) * 16 + h;
      Mha[((size_t)b * 2048 + q) * 1024 + c] = f32_to_bf16(oacc[nt].f[r] * inv);
    }
  }
}

// ---------------------------------------------------------------------------
extern "C" void kernel_launch(void* const* d_in, const int* in_sizes, int n_in,
                              void* d_out, int out_size, void* d_ws, size_t ws_size,
                              hipStream_t stream) {
  (void)in_sizes; (void)n_in; (void)out_size; (void)ws_size;
  const float* queries = (const float*)d_in[0];
  const float* keysp   = (const float*)d_in[1];
  const float* values  = (const float*)d_in[2];
  const int*   mask    = (const int*)d_in[3];
  const float* Wq = (const float*)d_in[4];
  const float* Wk = (const float*)d_in[5];
  const float* Wv = (const float*)d_in[6];
  const float* Wo = (const float*)d_in[7];

  const size_t PER = (size_t)4 * 16 * 2048 * 64;   // 8,388,608 bf16 = 16 MB
  uint16_t* Qp  = (uint16_t*)d_ws;
  uint16_t* Kp  = Qp + PER;
  uint16_t* VpT = Kp + PER;
  uint16_t* Mha = VpT + PER;                        // total ws use: 64 MB

  dim3 gGrid(8192 / BM, GN / BN);                   // 64 x 16
  gemm_wmma<0, 0><<<gGrid, 256, 0, stream>>>(queries, Wq, Qp);
  gemm_wmma<0, 0><<<gGrid, 256, 0, stream>>>(keysp,   Wk, Kp);
  gemm_wmma<0, 2><<<gGrid, 256, 0, stream>>>(values,  Wv, VpT);
  attn_wmma<<<dim3(128, 16, 4), 32, 0, stream>>>(Qp, Kp, VpT, mask, Mha);
  gemm_wmma<1, 3><<<gGrid, 256, 0, stream>>>(Mha, Wo, d_out);
}
